// LSTMModel_20744692040547
// MI455X (gfx1250) — matrix operand order repaired and checked by reading, not compiled, not run on hardware
//
#include <hip/hip_runtime.h>

// ---------------------------------------------------------------------------
// Fused persistent 2-layer LSTM + FC head for MI455X (gfx1250, wave32, WMMA).
// One wave per 16 batch rows; loops over all T steps with h/c state on-chip.
// GEMMs: v_wmma_f32_16x16x32_f16 (A = batch tile f16, B = weights f16,
// C/D = f32). Weights live in registers (32 fragments); h transposes go
// through a tiny LDS staging tile.
// ---------------------------------------------------------------------------

typedef __attribute__((ext_vector_type(16))) _Float16 v16h;
typedef __attribute__((ext_vector_type(8)))  _Float16 v8h;
typedef __attribute__((ext_vector_type(8)))  float    v8f;

#define HID 32
#define DIN 9

__device__ __forceinline__ float fast_rcp(float x) { return __builtin_amdgcn_rcpf(x); }
__device__ __forceinline__ float sigm(float x)     { return fast_rcp(1.0f + __expf(-x)); }
__device__ __forceinline__ float tanh_fast(float x){
    float e = __expf(2.0f * x);              // inf for large x -> result 1
    return 1.0f - 2.0f * fast_rcp(e + 1.0f); // -1 for very negative x
}

__device__ __forceinline__ v16h zero16() {
    v16h z;
#pragma unroll
    for (int i = 0; i < 16; ++i) z[i] = (_Float16)0.0f;
    return z;
}

// Load 16 consecutive f32 weights (one B-fragment lane slice) and convert.
__device__ __forceinline__ v16h load16(const float* __restrict__ p) {
    const float4* q = (const float4*)p;
    float4 a = q[0], b = q[1], c = q[2], d = q[3];
    v16h r;
    r[0]=(_Float16)a.x;  r[1]=(_Float16)a.y;  r[2]=(_Float16)a.z;  r[3]=(_Float16)a.w;
    r[4]=(_Float16)b.x;  r[5]=(_Float16)b.y;  r[6]=(_Float16)b.z;  r[7]=(_Float16)b.w;
    r[8]=(_Float16)c.x;  r[9]=(_Float16)c.y;  r[10]=(_Float16)c.z; r[11]=(_Float16)c.w;
    r[12]=(_Float16)d.x; r[13]=(_Float16)d.y; r[14]=(_Float16)d.z; r[15]=(_Float16)d.w;
    return r;
}

__global__ __launch_bounds__(32, 1) void lstm2_fused(
    const float* __restrict__ x,
    const float* __restrict__ Wih0, const float* __restrict__ Whh0,
    const float* __restrict__ bih0, const float* __restrict__ bhh0,
    const float* __restrict__ Wih1, const float* __restrict__ Whh1,
    const float* __restrict__ bih1, const float* __restrict__ bhh1,
    const float* __restrict__ Wfc,  const float* __restrict__ bfc,
    float* __restrict__ out, int B, int T)
{
    // h staging tiles: [0] = h of layer0, [1] = h of layer1. Row-major
    // [16 rows][32 cols] f16, 64B row stride. 2 KB total.
    __shared__ __align__(16) _Float16 lds[2][16 * HID];

    const int  lane = threadIdx.x;
    const bool loHalf = lane < 16;
    const int  mrow = lane & 15;              // batch row within tile / N col
    const int  b0   = blockIdx.x * 16;
    const int  brow = b0 + mrow;
    const int  brc  = (brow < B) ? brow : (B - 1);   // clamped for loads

    // ---- zero h state in LDS --------------------------------------------
    for (int i = lane; i < 2 * 16 * HID; i += 32) ((_Float16*)lds)[i] = (_Float16)0.0f;
    __syncthreads();

    // ---- biases: per-lane, per N-tile (same for all 8 rows of a D tile) --
    float bias0[8], bias1[8];
#pragma unroll
    for (int nt = 0; nt < 8; ++nt) {
        int n = nt * 16 + mrow;
        bias0[nt] = bih0[n] + bhh0[n];
        bias1[nt] = bih1[n] + bhh1[n];
    }

    // ---- weights -> register-resident B fragments ------------------------
    // B element (k, n): lanes 0-15 hold K=k0..k0+15 (k0 = ktile*32),
    // lanes 16-31 hold K=k0+16..k0+31, N = lane&15 + ntile*16.
    // wB[layer][ktile][ntile]
    v16h wB[2][2][8];
    const int k0 = loHalf ? 0 : 16;
#pragma unroll
    for (int nt = 0; nt < 8; ++nt) {
        int n = nt * 16 + mrow;
        // layer0 ktile0: x weights, only K=0..8 valid (zero-padded).
        v16h w00 = zero16();
        {
            const float* r = Wih0 + n * DIN;
            if (loHalf) {
#pragma unroll
                for (int i = 0; i < 8; ++i) w00[i] = (_Float16)r[i];
            } else {
                w00[0] = (_Float16)r[8];
            }
        }
        wB[0][0][nt] = w00;
        wB[0][1][nt] = load16(Whh0 + n * HID + k0);   // layer0 recurrent
        wB[1][0][nt] = load16(Wih1 + n * HID + k0);   // layer1 input (= h0)
        wB[1][1][nt] = load16(Whh1 + n * HID + k0);   // layer1 recurrent
    }

    // ---- helpers ---------------------------------------------------------
    // A-fragment of a 16x32 f16 tile stored row-major in lds[which].
    // lanes 0-15: elems 0-15 = K0-15 (row = lane)
    // lanes16-31: elems 0-15 = K16-31 (row = lane-16)
    auto fragA = [&](int which) -> v16h {
        const _Float16* p = &lds[which][mrow * HID + (loHalf ? 0 : 16)];
        v8h p0 = *(const v8h*)p;
        v8h p1 = *(const v8h*)(p + 8);
        return __builtin_shufflevector(p0, p1, 0,1,2,3,4,5,6,7,8,9,10,11,12,13,14,15);
    };

    // Raw x row fetch for step t (lanes 0-15: x[.,0..7]; lanes 16-31: x[.,8]).
    auto loadx = [&](int t, float* dst) {
        const float* row = x + ((size_t)brc * T + t) * DIN;
        if (loHalf) {
#pragma unroll
            for (int i = 0; i < 8; ++i) dst[i] = row[i];
        } else {
            dst[0] = row[8];
#pragma unroll
            for (int i = 1; i < 8; ++i) dst[i] = 0.0f;
        }
    };

    // One layer GEMM: acc[nt] = bias + a0*wk[0][nt] + a1*wk[1][nt]
    auto gemm = [&](const v16h& a0, const v16h& a1, const v16h (&wk)[2][8],
                    const float (&bias)[8], v8f (&acc)[8]) {
#pragma unroll
        for (int nt = 0; nt < 8; ++nt) {
            v8f cacc;
#pragma unroll
            for (int i = 0; i < 8; ++i) cacc[i] = bias[nt];
            cacc = __builtin_amdgcn_wmma_f32_16x16x32_f16(false, a0, false, wk[0][nt],
                                                          (short)0, cacc, false, false);
            cacc = __builtin_amdgcn_wmma_f32_16x16x32_f16(false, a1, false, wk[1][nt],
                                                          (short)0, cacc, false, false);
            acc[nt] = cacc;
        }
    };

    // LSTM pointwise + write h (f16) into staging tile `which`.
    // D layout: elem r of a tile <-> row (r + 8*hiHalf), col (mrow + 16*j).
    auto pointwise = [&](v8f (&acc)[8], v8f (&cst)[2], int which) {
        const int rofs = loHalf ? 0 : 8;
#pragma unroll
        for (int j = 0; j < 2; ++j) {
#pragma unroll
            for (int r = 0; r < 8; ++r) {
                float ig = sigm(acc[0 + j][r]);
                float fg = sigm(acc[2 + j][r]);
                float gg = tanh_fast(acc[4 + j][r]);
                float og = sigm(acc[6 + j][r]);
                float cc = fg * cst[j][r] + ig * gg;
                cst[j][r] = cc;
                float hh = og * tanh_fast(cc);
                lds[which][(r + rofs) * HID + (mrow + 16 * j)] = (_Float16)hh;
            }
        }
    };

    // ---- state -----------------------------------------------------------
    v8f c0[2], c1[2];
#pragma unroll
    for (int j = 0; j < 2; ++j)
#pragma unroll
        for (int r = 0; r < 8; ++r) { c0[j][r] = 0.0f; c1[j][r] = 0.0f; }

    float xr[8];
    loadx(0, xr);

    // ---- sequential time loop -------------------------------------------
    for (int t = 0; t < T; ++t) {
        // layer0 A fragments: x tile (built in regs) + previous h0 (LDS)
        v16h ax = zero16();
#pragma unroll
        for (int i = 0; i < 8; ++i) ax[i] = (_Float16)xr[i];
        v16h ah0 = fragA(0);

        v8f acc[8];
        gemm(ax, ah0, wB[0], bias0, acc);

        // prefetch next step's x while the GEMM/pointwise run
        int tn = (t + 1 < T) ? (t + 1) : t;
        float xn[8];
        loadx(tn, xn);

        pointwise(acc, c0, 0);        // writes new h0 into lds[0]
        __syncthreads();

        // layer1: A = [h0(t), h1(t-1)]
        v16h a10 = fragA(0);
        v16h a11 = fragA(1);
        gemm(a10, a11, wB[1], bias1, acc);
        pointwise(acc, c1, 1);        // writes new h1 into lds[1]
        __syncthreads();

#pragma unroll
        for (int i = 0; i < 8; ++i) xr[i] = xn[i];
    }

    // ---- FC head from final h1 (lds[1]) ---------------------------------
    if (loHalf && brow < B) {
#pragma unroll
        for (int j = 0; j < 3; ++j) {
            float s = bfc[j];
#pragma unroll
            for (int k = 0; k < HID; ++k)
                s += (float)lds[1][mrow * HID + k] * Wfc[j * HID + k];
            out[brow * 3 + j] = s;
        }
    }
}

extern "C" void kernel_launch(void* const* d_in, const int* in_sizes, int n_in,
                              void* d_out, int out_size, void* d_ws, size_t ws_size,
                              hipStream_t stream) {
    const float* x    = (const float*)d_in[0];
    const float* Wih0 = (const float*)d_in[1];
    const float* Whh0 = (const float*)d_in[2];
    const float* bih0 = (const float*)d_in[3];
    const float* bhh0 = (const float*)d_in[4];
    const float* Wih1 = (const float*)d_in[5];
    const float* Whh1 = (const float*)d_in[6];
    const float* bih1 = (const float*)d_in[7];
    const float* bhh1 = (const float*)d_in[8];
    const float* Wfc  = (const float*)d_in[9];
    const float* bfc  = (const float*)d_in[10];

    const int B = out_size / 3;                 // 8192
    const int T = in_sizes[0] / (B * DIN);      // 365
    const int blocks = (B + 15) / 16;           // 512 waves, one per 16 rows

    lstm2_fused<<<blocks, 32, 0, stream>>>(x, Wih0, Whh0, bih0, bhh0,
                                           Wih1, Whh1, bih1, bhh1,
                                           Wfc, bfc, (float*)d_out, B, T);
    (void)d_ws; (void)ws_size; (void)n_in;
}
